// KNN_80513456931114
// MI455X (gfx1250) — matrix operand order repaired and checked by reading, not compiled, not run on hardware
//
#include <hip/hip_runtime.h>
#include <hip/hip_bf16.h>

// ---------------- problem constants ----------------
#define N_TRAIN   500000
#define BATCH     512
#define D_RAW     3072
#define D_PROJ    30
#define D_PAD     32          // K padded to one 16x16x32 WMMA
#define TOPK      15
#define N_CLASSES 10

// slice geometry: 250 * 2000 == 500000, and 2000 % 16 == 0 (no row duplication)
#define N_SLICES  250
#define SLICE_LEN 2000
#define Q_TILES   (BATCH / 16)            // 32 query tiles of 16

// main-kernel tiling: block = 4 waves sharing one data slice,
// each wave owns 2 query tiles  ->  8 query tiles per data stream.
#define QT_PER_WAVE     2
#define WAVES_PER_BLOCK 4
#define QT_PER_BLOCK    (QT_PER_WAVE * WAVES_PER_BLOCK)   // 8
#define QGROUPS         (Q_TILES / QT_PER_BLOCK)          // 4

#define CAND_PER_Q (N_SLICES * 2 * TOPK)  // 7500 candidates per query

typedef __attribute__((ext_vector_type(16))) _Float16 v16h;
typedef __attribute__((ext_vector_type(8)))  _Float16 v8h;
typedef __attribute__((ext_vector_type(8)))  float    v8f;
typedef __attribute__((ext_vector_type(4)))  float    v4f;

#define BIGF 3.4e38f

// Register-resident top-K insertion: fully unrolled so bd/bi stay in VGPRs.
__device__ __forceinline__ void topk_insert(float (&bd)[TOPK], int (&bi)[TOPK],
                                            float& worst, float d, int idx) {
    int wp = 0; float wv = bd[0];
#pragma unroll
    for (int k = 1; k < TOPK; ++k) { if (bd[k] > wv) { wv = bd[k]; wp = k; } }
#pragma unroll
    for (int k = 0; k < TOPK; ++k) { if (k == wp) { bd[k] = d; bi[k] = idx; } }
    wv = bd[0];
#pragma unroll
    for (int k = 1; k < TOPK; ++k) wv = fmaxf(wv, bd[k]);
    worst = wv;
}

__device__ __forceinline__ float min8(const float (&s)[8]) {
    return fminf(fminf(fminf(s[0], s[1]), fminf(s[2], s[3])),
                 fminf(fminf(s[4], s[5]), fminf(s[6], s[7])));
}

// ---------------------------------------------------------------------------
// Kernel 1: per-query centering, L2-normalization, projection to 30-d.
// Emits f16 px padded to 32 (query-major, 64B rows) + f32 squared norms.
// One block (256 threads) per query row.
// ---------------------------------------------------------------------------
__global__ void prep_queries(const float* __restrict__ x,
                             const float* __restrict__ proj,
                             _Float16* __restrict__ px16,
                             float* __restrict__ qnorm) {
    __shared__ float smem[256 * D_PROJ];   // reused: first 3072 = x row, then 256x30 partials
    __shared__ float red[256];
    const int q = blockIdx.x;
    const int t = threadIdx.x;
    const float* xr = x + (size_t)q * D_RAW;

    // mean
    float s = 0.f;
    for (int i = t; i < D_RAW; i += 256) { float v = xr[i]; smem[i] = v; s += v; }
    red[t] = s; __syncthreads();
    for (int o = 128; o > 0; o >>= 1) { if (t < o) red[t] += red[t + o]; __syncthreads(); }
    const float mean = red[0] * (1.0f / D_RAW);
    __syncthreads();

    // center + sum of squares
    float ss = 0.f;
    for (int i = t; i < D_RAW; i += 256) { float v = smem[i] - mean; smem[i] = v; ss += v * v; }
    red[t] = ss; __syncthreads();
    for (int o = 128; o > 0; o >>= 1) { if (t < o) red[t] += red[t + o]; __syncthreads(); }
    const float inv = rsqrtf(red[0]);
    __syncthreads();

    // projection: each thread accumulates all 30 outputs over its i-stride
    float acc[D_PROJ];
#pragma unroll
    for (int j = 0; j < D_PROJ; ++j) acc[j] = 0.f;
    for (int i = t; i < D_RAW; i += 256) {
        const float v = smem[i] * inv;
        const float* pr = proj + (size_t)i * D_PROJ;
#pragma unroll
        for (int j = 0; j < D_PROJ; ++j) acc[j] = fmaf(v, pr[j], acc[j]);
    }
    __syncthreads();                       // done reading smem as x-row
#pragma unroll
    for (int j = 0; j < D_PROJ; ++j) smem[t * D_PROJ + j] = acc[j];
    __syncthreads();

    if (t < D_PROJ) {
        float sum = 0.f;
        for (int tt = 0; tt < 256; ++tt) sum += smem[tt * D_PROJ + t];
        px16[(size_t)q * D_PAD + t] = (_Float16)sum;
        red[t] = sum * sum;
    } else if (t < D_PAD) {
        px16[(size_t)q * D_PAD + t] = (_Float16)0.f;   // K padding
    }
    __syncthreads();
    if (t == 0) {
        float n2 = 0.f;
        for (int j = 0; j < D_PROJ; ++j) n2 += red[j];
        qnorm[q] = n2;
    }
}

// ---------------------------------------------------------------------------
// Kernel 2: pack train data rows to f16 padded to 32 (64B rows) + f32 norms.
// One thread per train row.
// ---------------------------------------------------------------------------
__global__ void prep_data(const float* __restrict__ data,
                          _Float16* __restrict__ data16,
                          float* __restrict__ dnorm) {
    const int r = blockIdx.x * blockDim.x + threadIdx.x;
    if (r >= N_TRAIN) return;
    const float* dr = data + (size_t)r * D_PROJ;
    float n2 = 0.f;
    _Float16 tmp[D_PAD];
#pragma unroll
    for (int j = 0; j < D_PROJ; ++j) { float v = dr[j]; n2 += v * v; tmp[j] = (_Float16)v; }
    tmp[30] = (_Float16)0.f; tmp[31] = (_Float16)0.f;
    _Float16* o = data16 + (size_t)r * D_PAD;       // 64-byte aligned row
#pragma unroll
    for (int c = 0; c < 4; ++c) *(v8h*)(o + c * 8) = *(const v8h*)(tmp + c * 8);
    dnorm[r] = n2;
}

// ---------------------------------------------------------------------------
// Kernel 3: fused WMMA distance + streaming per-lane top-K.
// Block = 4 waves, all walking the SAME data slice (A/dnorm loads hit WGP$,
// one L2 stream per block).  Each wave owns 2 query tiles: per data tile it
// issues 2 independent v_wmma_f32_16x16x32_f16 back-to-back.
// Selection key: s = ||d||^2 - 2*q.d  (qn added at merge; same ordering).
// Fast path per C-vector: 8 fma + min-tree + ONE compare/branch; the
// per-element insert scan runs only when the group minimum beats `worst`.
// ---------------------------------------------------------------------------
__global__ void knn_topk(const _Float16* __restrict__ px16,
                         const _Float16* __restrict__ data16,
                         const float*    __restrict__ dnorm,
                         float* __restrict__ cand_d,
                         int*   __restrict__ cand_i) {
    const int lane   = threadIdx.x & 31;
    const int wave   = threadIdx.x >> 5;
    const int qgroup = blockIdx.x % QGROUPS;
    const int slice  = blockIdx.x / QGROUPS;
    const int qt0    = qgroup * QT_PER_BLOCK + wave * QT_PER_WAVE;
    const int hi     = lane >> 4;            // 0: lanes 0-15, 1: lanes 16-31

    const int q0 = (qt0 + 0) * 16 + (lane & 15);
    const int q1 = (qt0 + 1) * 16 + (lane & 15);
    // B operands: query q, K halves [0..15] (lo lanes) or [16..31] (hi lanes)
    const v16h b0 = *(const v16h*)(px16 + (size_t)q0 * D_PAD + hi * 16);
    const v16h b1 = *(const v16h*)(px16 + (size_t)q1 * D_PAD + hi * 16);

    float bd0[TOPK], bd1[TOPK]; int bi0[TOPK], bi1[TOPK];
#pragma unroll
    for (int k = 0; k < TOPK; ++k) {
        bd0[k] = BIGF; bi0[k] = 0; bd1[k] = BIGF; bi1[k] = 0;
    }
    float worst0 = BIGF, worst1 = BIGF;

    const int base   = slice * SLICE_LEN;
    const int rowoff = hi * 8;               // this lane's 8 data rows within tile

    for (int tb = 0; tb < SLICE_LEN; tb += 16) {
        const int d0 = base + tb;
        // A operand (16x32, this kernel's contiguous K order, not the ISA table order): lane m = lane&15 is matrix row m;
        // lo lanes take K{0..15}, hi lanes K{16..31} (matches B).
        const _Float16* arow = data16 + (size_t)(d0 + (lane & 15)) * D_PAD;
        v16h a = *(const v16h*)(arow + hi * 16);

        // prefetch next tile's rows (global_prefetch_b8); stream shared by block
        __builtin_prefetch((const void*)(arow + 16 * D_PAD), 0, 3);

        v8f c0 = {}, c1 = {};
        c0 = __builtin_amdgcn_wmma_f32_16x16x32_f16(
                false, a, false, b0, (short)0, c0, false, false);
        c1 = __builtin_amdgcn_wmma_f32_16x16x32_f16(
                false, a, false, b1, (short)0, c1, false, false);

        const float* dnp = dnorm + d0 + rowoff;      // 32B aligned
        const v4f dn0 = *(const v4f*)(dnp);
        const v4f dn1 = *(const v4f*)(dnp + 4);
        const float dnv[8] = { dn0[0], dn0[1], dn0[2], dn0[3],
                               dn1[0], dn1[1], dn1[2], dn1[3] };

        float s0v[8], s1v[8];
#pragma unroll
        for (int r = 0; r < 8; ++r) s0v[r] = fmaf(-2.0f, c0[r], dnv[r]);
#pragma unroll
        for (int r = 0; r < 8; ++r) s1v[r] = fmaf(-2.0f, c1[r], dnv[r]);
        const float m0 = min8(s0v);
        const float m1 = min8(s1v);

        if (m0 < worst0) {
#pragma unroll
            for (int r = 0; r < 8; ++r)
                if (s0v[r] < worst0)
                    topk_insert(bd0, bi0, worst0, s0v[r], d0 + rowoff + r);
        }
        if (m1 < worst1) {
#pragma unroll
            for (int r = 0; r < 8; ++r)
                if (s1v[r] < worst1)
                    topk_insert(bd1, bi1, worst1, s1v[r], d0 + rowoff + r);
        }
    }

    // candidate layout: [query][slice][half-lane][TOPK]
    const size_t cb0 = (((size_t)q0 * N_SLICES + slice) * 2 + hi) * TOPK;
    const size_t cb1 = (((size_t)q1 * N_SLICES + slice) * 2 + hi) * TOPK;
#pragma unroll
    for (int k = 0; k < TOPK; ++k) {
        cand_d[cb0 + k] = bd0[k]; cand_i[cb0 + k] = bi0[k];
        cand_d[cb1 + k] = bd1[k]; cand_i[cb1 + k] = bi1[k];
    }
}

// ---------------------------------------------------------------------------
// Kernel 4: per-query merge of 7500 candidates -> global top-15 ->
// out[q][c] = log( sum_k exp(-sqrt(qn + s_k)) * labels[idx_k][c] ).
// One block (256 threads) per query.
// ---------------------------------------------------------------------------
__global__ void knn_merge(const float* __restrict__ cand_d,
                          const int*   __restrict__ cand_i,
                          const float* __restrict__ qnorm,
                          const float* __restrict__ labels,
                          float* __restrict__ out) {
    __shared__ float sd[256 * TOPK];
    __shared__ int   si[256 * TOPK];
    __shared__ float rv[256];
    __shared__ int   rp[256];
    __shared__ float seld[TOPK];
    __shared__ int   seli[TOPK];

    const int q = blockIdx.x;
    const int t = threadIdx.x;
    const float* cd = cand_d + (size_t)q * CAND_PER_Q;
    const int*   ci = cand_i + (size_t)q * CAND_PER_Q;

    float bd[TOPK]; int bi[TOPK];
#pragma unroll
    for (int k = 0; k < TOPK; ++k) { bd[k] = BIGF; bi[k] = 0; }
    float worst = BIGF;
    for (int i = t; i < CAND_PER_Q; i += 256) {
        const float d = cd[i];
        if (d < worst) topk_insert(bd, bi, worst, d, ci[i]);
    }
#pragma unroll
    for (int k = 0; k < TOPK; ++k) { sd[t * TOPK + k] = bd[k]; si[t * TOPK + k] = bi[k]; }
    __syncthreads();

    // 15 rounds of block-wide min with invalidation
    for (int k = 0; k < TOPK; ++k) {
        float mv = BIGF; int mp = -1;
#pragma unroll
        for (int j = 0; j < TOPK; ++j) {
            const float v = sd[t * TOPK + j];
            if (v < mv) { mv = v; mp = t * TOPK + j; }
        }
        rv[t] = mv; rp[t] = mp; __syncthreads();
        for (int o = 128; o > 0; o >>= 1) {
            if (t < o && rv[t + o] < rv[t]) { rv[t] = rv[t + o]; rp[t] = rp[t + o]; }
            __syncthreads();
        }
        if (t == 0 && rp[0] >= 0) {
            seld[k] = rv[0]; seli[k] = si[rp[0]]; sd[rp[0]] = BIGF;
        }
        __syncthreads();
    }

    if (t < N_CLASSES) {
        const float qn = qnorm[q];
        float acc = 0.f;
        for (int k = 0; k < TOPK; ++k) {
            const float d = sqrtf(fmaxf(qn + seld[k], 0.f));   // d^2 = qn + s
            acc += expf(-d) * labels[(size_t)seli[k] * N_CLASSES + t];
        }
        out[(size_t)q * N_CLASSES + t] = logf(acc);
    }
}

// ---------------------------------------------------------------------------
extern "C" void kernel_launch(void* const* d_in, const int* in_sizes, int n_in,
                              void* d_out, int out_size, void* d_ws, size_t ws_size,
                              hipStream_t stream) {
    const float* x         = (const float*)d_in[0];   // [512, 3072]
    const float* projector = (const float*)d_in[1];   // [3072, 30]
    const float* data      = (const float*)d_in[2];   // [500000, 30]
    const float* labels    = (const float*)d_in[3];   // [500000, 10]
    float* out = (float*)d_out;                       // [512, 10]

    char* ws = (char*)d_ws;
    size_t off = 0;
    auto carve = [&](size_t bytes) -> void* {
        void* p = ws + off;
        off = (off + bytes + 255) & ~(size_t)255;
        return p;
    };
    _Float16* px16   = (_Float16*)carve((size_t)BATCH * D_PAD * sizeof(_Float16));
    float*    qnorm  = (float*)   carve((size_t)BATCH * sizeof(float));
    _Float16* data16 = (_Float16*)carve((size_t)N_TRAIN * D_PAD * sizeof(_Float16));
    float*    dnorm  = (float*)   carve((size_t)N_TRAIN * sizeof(float));
    float*    cand_d = (float*)   carve((size_t)BATCH * CAND_PER_Q * sizeof(float));
    int*      cand_i = (int*)     carve((size_t)BATCH * CAND_PER_Q * sizeof(int));
    (void)ws_size; (void)in_sizes; (void)n_in; (void)out_size;

    prep_queries<<<BATCH, 256, 0, stream>>>(x, projector, px16, qnorm);
    prep_data<<<(N_TRAIN + 255) / 256, 256, 0, stream>>>(data, data16, dnorm);

    // 1000 blocks x 4 waves; each block streams one slice for 8 query tiles
    knn_topk<<<QGROUPS * N_SLICES, 32 * WAVES_PER_BLOCK, 0, stream>>>(
        px16, data16, dnorm, cand_d, cand_i);
    knn_merge<<<BATCH, 256, 0, stream>>>(cand_d, cand_i, qnorm, labels, out);
}
